// DynamicScaleConv_45835890983316
// MI455X (gfx1250) — matrix operand order repaired and checked by reading, not compiled, not run on hardware
//
#include <hip/hip_runtime.h>

// ---------------------------------------------------------------------------
// Types for CDNA5 WMMA (wave32): D(16x16,f32) = A(16x32,bf16) x B(32x16,bf16) + C
// ---------------------------------------------------------------------------
typedef __attribute__((ext_vector_type(16))) __bf16 v16bf;
typedef __attribute__((ext_vector_type(8)))  float  v8f;

union AFrag { v16bf v; unsigned u[8]; uint4 q[2]; };

// f32 pair -> packed bf16 u32 (RNE). Prefer hardware v_cvt_pk_bf16_f32.
static __device__ __forceinline__ unsigned pack_bf16(float lo, float hi) {
#if __has_builtin(__builtin_amdgcn_cvt_pk_bf16_f32)
    typedef __attribute__((ext_vector_type(2))) __bf16 v2bf;
    v2bf r = __builtin_amdgcn_cvt_pk_bf16_f32(lo, hi);
    return __builtin_bit_cast(unsigned, r);
#else
    unsigned a = __float_as_uint(lo);
    unsigned b = __float_as_uint(hi);
    a = (a + 0x7FFFu + ((a >> 16) & 1u)) >> 16;
    b = (b + 0x7FFFu + ((b >> 16) & 1u)) >> 16;
    return (a & 0xFFFFu) | (b << 16);
#endif
}

// ---------------------------------------------------------------------------
// Problem constants:  B=8, C=256, H=W=128, K = C*9 = 2304 (ordered tap*256+c)
// ---------------------------------------------------------------------------
#define NB    8
#define NC    256
#define NHW   128
#define NKT   72          // 2304/32 K-tiles of 32
#define WPACK_U32 294912  // (2304/2)*256

// ---------------------------------------------------------------------------
// Kernel 1: pack weights into bf16-pair GEMM-B layout, zero pooling sums.
//   w_pack[kp*256 + n] = bf16pair( w[n][c][r][s], w[n][c+1][r][s] ),
//   k = (r*3+s)*256 + c, kp = k/2.
// ---------------------------------------------------------------------------
__global__ void __launch_bounds__(256)
prep_kernel(const float* __restrict__ wd, unsigned* __restrict__ wpack,
            float* __restrict__ sums) {
    int idx = blockIdx.x * 256 + threadIdx.x;      // < 294912 exactly
    int kp  = idx >> 8;
    int n   = idx & 255;
    int k0  = kp << 1;
    int tap = k0 >> 8;                              // 0..8  (r*3+s)
    int c   = k0 & 255;                             // even
    const float* src = wd + (size_t)((n << 8) + c) * 9 + tap;
    wpack[idx] = pack_bf16(src[0], src[9]);         // c and c+1, same tap
    if (idx < NB * NC) sums[idx] = 0.0f;            // must re-zero every call
}

// ---------------------------------------------------------------------------
// Kernel 2: implicit-GEMM conv via WMMA bf16.
//   Block: 256 threads (8 wave32), tile M=128 (one image row) x N=128 channels.
//   Wave (wm 0..3, wn 0..1) owns 32x64 -> 8 accumulators of 16x16.
//   K loop: 72 tiles of 32; each K-tile lies within one (r,s) tap.
//   LDS layouts pair-packed + padded to 80B rows so all fragment traffic is b128.
// ---------------------------------------------------------------------------
__global__ void __launch_bounds__(256)
conv_wmma_kernel(const float* __restrict__ x, const unsigned* __restrict__ wpack,
                 const float* __restrict__ bdef, float* __restrict__ yout,
                 float* __restrict__ sums) {
    __shared__ unsigned a_lds[2][128][20];   // [buf][pixel-row][kpair 0..15] pad->20
    __shared__ unsigned b_t[2][128][20];     // [buf][n 0..127][kpair 0..15] pad->20
    __shared__ float    red[128];

    const int t      = threadIdx.x;
    const int lane   = t & 31;
    const int wv     = t >> 5;
    const int wave_m = wv & 3;
    const int wave_n = wv >> 2;
    const int lrow   = lane & 15;
    const int lhalf  = lane >> 4;

    const int mt   = blockIdx.x;        // 0..1023 : pixel-row tile (one image row)
    const int n0   = blockIdx.y << 7;   // 0 or 128: channel tile base
    const int bImg = mt >> 7;
    const int hRow = mt & 127;

    const int arow  = t & 127;          // w coordinate staged by this thread
    const int khalf = t >> 7;           // which 16 of the 32 k-values

    v8f acc[2][4] = {};

    // ---- prologue: stage K-tile 0 into buffer 0 ----
    {
        const int hh = hRow - 1, ww = arow - 1;     // tap 0: dr = ds = -1
        const bool valid = ((unsigned)hh < 128u) & ((unsigned)ww < 128u);
        float aval[16];
        if (valid) {
            const float* xb = x + ((((size_t)bImg << 8) + (khalf << 4)) << 14)
                                + (hh << 7) + ww;
            #pragma unroll
            for (int i = 0; i < 16; ++i) aval[i] = xb[(size_t)i << 14];
        } else {
            #pragma unroll
            for (int i = 0; i < 16; ++i) aval[i] = 0.0f;
        }
        uint4* adst = reinterpret_cast<uint4*>(&a_lds[0][arow][0]);
        adst[khalf * 2] = uint4{pack_bf16(aval[0], aval[1]), pack_bf16(aval[2], aval[3]),
                                pack_bf16(aval[4], aval[5]), pack_bf16(aval[6], aval[7])};
        adst[khalf * 2 + 1] = uint4{pack_bf16(aval[8],  aval[9]),  pack_bf16(aval[10], aval[11]),
                                    pack_bf16(aval[12], aval[13]), pack_bf16(aval[14], aval[15])};
        #pragma unroll
        for (int j = 0; j < 8; ++j) {
            int idx = t + (j << 8);
            int n   = idx & 127, kp = idx >> 7;
            b_t[0][n][kp] = wpack[(kp << 8) + n0 + n];
        }
    }
    __syncthreads();

    // ---- main K loop, double buffered ----
    for (int kt = 0; kt < NKT; ++kt) {
        const int  cur     = kt & 1;
        const bool havnext = (kt + 1) < NKT;
        float aval[16];
        unsigned bval[8];
        bool valid = false;

        // prefetch x row needed two K-tiles ahead (global_prefetch_b8)
        if (kt + 2 < NKT) {
            const int kn2  = kt + 2;
            const int tap2 = kn2 >> 3;
            const int c02  = (kn2 & 7) << 5;
            const int hh2  = hRow + (tap2 / 3) - 1;
            const int ww2  = arow + (tap2 % 3) - 1;
            if (((unsigned)hh2 < 128u) & ((unsigned)ww2 < 128u))
                __builtin_prefetch(x + ((((size_t)bImg << 8) + c02 + (khalf << 4)) << 14)
                                     + (hh2 << 7) + ww2, 0, 3);
        }

        if (havnext) {
            const int kn  = kt + 1;
            const int tap = kn >> 3;                 // 0..8
            const int c0  = (kn & 7) << 5;
            const int hh  = hRow + (tap / 3) - 1;
            const int ww  = arow + (tap % 3) - 1;
            valid = ((unsigned)hh < 128u) & ((unsigned)ww < 128u);
            if (valid) {
                const float* xb = x + ((((size_t)bImg << 8) + c0 + (khalf << 4)) << 14)
                                    + (hh << 7) + ww;
                #pragma unroll
                for (int i = 0; i < 16; ++i) aval[i] = xb[(size_t)i << 14];
            }
            #pragma unroll
            for (int j = 0; j < 8; ++j) {
                int idx = t + (j << 8);
                bval[j] = wpack[(((kn << 4) + (idx >> 7)) << 8) + n0 + (idx & 127)];
            }
        }

        // Fragment loads from LDS: all 128-bit.
        AFrag af[2];
        AFrag bf[4];
        #pragma unroll
        for (int ms = 0; ms < 2; ++ms) {
            const int rb = (wave_m << 5) + (ms << 4) + lrow;
            const uint4* ar = reinterpret_cast<const uint4*>(&a_lds[cur][rb][0]);
            af[ms].q[0] = ar[lhalf];           // kpairs 4*lhalf   .. +3  (v0..v3)
            af[ms].q[1] = ar[2 + lhalf];       // kpairs 8+4*lhalf .. +3  (v4..v7)
        }
        #pragma unroll
        for (int ns = 0; ns < 4; ++ns) {
            const int nc = (wave_n << 6) + (ns << 4) + lrow;
            const uint4* br = reinterpret_cast<const uint4*>(&b_t[cur][nc][0]);
            bf[ns].q[0] = br[lhalf];           // kpairs 4*lhalf   .. +3
            bf[ns].q[1] = br[2 + lhalf];       // kpairs 8+4*lhalf .. +3
        }

        #pragma unroll
        for (int ms = 0; ms < 2; ++ms)
            #pragma unroll
            for (int ns = 0; ns < 4; ++ns)
                acc[ms][ns] = __builtin_amdgcn_wmma_f32_16x16x32_bf16(
                    false, af[ms].v, false, bf[ns].v,
                    (short)0, acc[ms][ns], false, false);

        if (havnext) {
            const int nxt = cur ^ 1;
            if (!valid) {
                #pragma unroll
                for (int i = 0; i < 16; ++i) aval[i] = 0.0f;
            }
            uint4* adst = reinterpret_cast<uint4*>(&a_lds[nxt][arow][0]);
            adst[khalf * 2] = uint4{pack_bf16(aval[0], aval[1]), pack_bf16(aval[2], aval[3]),
                                    pack_bf16(aval[4], aval[5]), pack_bf16(aval[6], aval[7])};
            adst[khalf * 2 + 1] = uint4{pack_bf16(aval[8],  aval[9]),  pack_bf16(aval[10], aval[11]),
                                        pack_bf16(aval[12], aval[13]), pack_bf16(aval[14], aval[15])};
            #pragma unroll
            for (int j = 0; j < 8; ++j) {
                int idx = t + (j << 8);
                b_t[nxt][idx & 127][idx >> 7] = bval[j];
            }
        }
        __syncthreads();
    }

    // ---- epilogue: bias add, vectorized y store, fused pooling reduce ----
    if (t < 128) red[t] = 0.0f;
    __syncthreads();

    #pragma unroll
    for (int ns = 0; ns < 4; ++ns) {
        const int nl = (wave_n << 6) + (ns << 4) + lrow;    // 0..127
        const int ng = n0 + nl;                             // global channel
        const float bias = bdef[ng];
        float s_local = 0.0f;
        #pragma unroll
        for (int ms = 0; ms < 2; ++ms) {
            const int wbase = (wave_m << 5) + (ms << 4) + (lhalf << 3);  // w coord
            float* yrow = yout + ((((size_t)bImg << 8) + ng) << 14) + (hRow << 7);
            float4 s0, s1;
            s0.x = acc[ms][ns][0] + bias; s0.y = acc[ms][ns][1] + bias;
            s0.z = acc[ms][ns][2] + bias; s0.w = acc[ms][ns][3] + bias;
            s1.x = acc[ms][ns][4] + bias; s1.y = acc[ms][ns][5] + bias;
            s1.z = acc[ms][ns][6] + bias; s1.w = acc[ms][ns][7] + bias;
            *reinterpret_cast<float4*>(yrow + wbase)     = s0;
            *reinterpret_cast<float4*>(yrow + wbase + 4) = s1;
            s_local += s0.x + s0.y + s0.z + s0.w + s1.x + s1.y + s1.z + s1.w;
        }
        atomicAdd(&red[nl], s_local);
    }
    __syncthreads();
    if (t < 128) atomicAdd(&sums[(bImg << 8) + n0 + t], red[t]);
}

// ---------------------------------------------------------------------------
// Kernel 3: SE head: pool -> FC(256->64)+ReLU -> FC(64->256) -> sigmoid
// ---------------------------------------------------------------------------
__global__ void __launch_bounds__(256)
se_kernel(const float* __restrict__ sums,
          const float* __restrict__ w1, const float* __restrict__ b1,
          const float* __restrict__ w2, const float* __restrict__ b2,
          float* __restrict__ scale) {
    __shared__ float p[NB * NC];
    __shared__ float hbuf[NB * 64];
    const int t = threadIdx.x;
    const float inv = 1.0f / (float)(NHW * NHW);
    for (int i = t; i < NB * NC; i += 256) p[i] = sums[i] * inv;
    __syncthreads();
    for (int o = t; o < NB * 64; o += 256) {
        const int b = o >> 6, j = o & 63;
        float d = b1[j];
        const float* pr = p + (b << 8);
        const float* wr = w1 + (j << 8);
        for (int c = 0; c < NC; ++c) d += pr[c] * wr[c];
        hbuf[o] = d > 0.0f ? d : 0.0f;
    }
    __syncthreads();
    for (int o = t; o < NB * NC; o += 256) {
        const int b = o >> 8, c = o & 255;
        float d = b2[c];
        const float* hr = hbuf + (b << 6);
        const float* wr = w2 + (c << 6);
        for (int j = 0; j < 64; ++j) d += hr[j] * wr[j];
        scale[o] = 1.0f / (1.0f + __expf(-d));
    }
}

// ---------------------------------------------------------------------------
// Kernel 4: in-place y *= scale[b,c], float4 vectorized.
// ---------------------------------------------------------------------------
__global__ void __launch_bounds__(256)
apply_scale_kernel(float* __restrict__ y, const float* __restrict__ scale) {
    const size_t e = ((size_t)blockIdx.x * 256 + threadIdx.x) << 2;
    float4 v = *reinterpret_cast<float4*>(y + e);
    const float s = scale[(unsigned)(e >> 14)];   // (b*256 + c)
    v.x *= s; v.y *= s; v.z *= s; v.w *= s;
    *reinterpret_cast<float4*>(y + e) = v;
}

// ---------------------------------------------------------------------------
// Launch
// ---------------------------------------------------------------------------
extern "C" void kernel_launch(void* const* d_in, const int* in_sizes, int n_in,
                              void* d_out, int out_size, void* d_ws, size_t ws_size,
                              hipStream_t stream) {
    (void)in_sizes; (void)n_in; (void)out_size; (void)ws_size;
    const float* x  = (const float*)d_in[0];
    const float* wd = (const float*)d_in[1];
    const float* bd = (const float*)d_in[2];
    const float* w1 = (const float*)d_in[3];
    const float* b1 = (const float*)d_in[4];
    const float* w2 = (const float*)d_in[5];
    const float* b2 = (const float*)d_in[6];
    float* y = (float*)d_out;

    unsigned* wpack = (unsigned*)d_ws;                                   // 1.18 MB
    float*    sums  = (float*)((char*)d_ws + (size_t)WPACK_U32 * 4);     // 8 KB
    float*    scale = sums + NB * NC;                                    // 8 KB

    prep_kernel<<<WPACK_U32 / 256, 256, 0, stream>>>(wd, wpack, sums);
    conv_wmma_kernel<<<dim3(NB * NHW * NHW / 128, NC / 128), 256, 0, stream>>>(
        x, wpack, bd, y, sums);
    se_kernel<<<1, 256, 0, stream>>>(sums, w1, b1, w2, b2, scale);
    apply_scale_kernel<<<(NB * NC * NHW * NHW) / (4 * 256), 256, 0, stream>>>(y, scale);
}
